// VectorQuantizer_51573967290740
// MI455X (gfx1250) — matrix operand order repaired and checked by reading, not compiled, not run on hardware
//
#include <hip/hip_runtime.h>
#include <hip/hip_bf16.h>

typedef unsigned int  u32;
typedef unsigned short u16;

typedef __attribute__((ext_vector_type(16))) __bf16 v16bf;
typedef __attribute__((ext_vector_type(8)))  float  v8f;

// ---------------- constants for this problem ----------------
#define BATCH 8
#define CH    256
#define HWS   4096          // 64*64
#define NPOS  32768         // BATCH*HWS
#define INTER 1024
#define KCB   1024
#define PA    48            // LDS row pitch in halfs (96B, 16B-aligned)

// 1 = use gfx1250 async VMEM->LDS (global_load_async_to_lds_b128 + s_wait_asynccnt)
// 0 = synchronous global->VGPR->LDS staging fallback
#define USE_ASYNC 1

// ---------------- helpers ----------------
__device__ __forceinline__ u16 f2bf(float f) {
  u32 u = __builtin_bit_cast(u32, f);
  u32 r = u + 0x7FFFu + ((u >> 16) & 1u);
  return (u16)(r >> 16);
}
__device__ __forceinline__ float bf2f(u16 h) {
  u32 u = ((u32)h) << 16;
  return __builtin_bit_cast(float, u);
}

union Frag { v16bf bf; uint4 q[2]; };
union H8   { u16 u[8]; uint4 q; };
union H4   { u16 u[4]; uint2 q; };

// stage 32 contiguous bytes: global gsrc -> LDS ldst
__device__ __forceinline__ void stage32(const u16* __restrict__ gsrc, u16* ldst) {
#if USE_ASYNC
  // VDST = LDS byte address (low 32 bits of the generic LDS pointer),
  // VADDR = 64-bit global address; inst offset applies to both sides.
  u32 l = (u32)(uintptr_t)ldst;
  unsigned long long g = (unsigned long long)(uintptr_t)gsrc;
  asm volatile("global_load_async_to_lds_b128 %0, %1, off" :: "v"(l), "v"(g) : "memory");
  asm volatile("global_load_async_to_lds_b128 %0, %1, off offset:16" :: "v"(l), "v"(g) : "memory");
#else
  uint4 a0 = ((const uint4*)gsrc)[0];
  uint4 a1 = ((const uint4*)gsrc)[1];
  *(uint4*)(ldst)     = a0;
  *(uint4*)(ldst + 8) = a1;
#endif
}
__device__ __forceinline__ void stage_wait() {
#if USE_ASYNC
  asm volatile("s_wait_asynccnt 0" ::: "memory");
#endif
}

// ---------------- small utility kernels ----------------
__global__ __launch_bounds__(256) void k_zero_f32(float* p, int n) {
  int i = blockIdx.x * 256 + threadIdx.x;
  if (i < n) p[i] = 0.0f;
}

// out[n*K + k] = bf16(in[k*N + n])   (weight transpose + convert)
__global__ __launch_bounds__(256) void k_convT(const float* __restrict__ in,
                                               u16* __restrict__ out, int K, int N) {
  int i = blockIdx.x * 256 + threadIdx.x;
  if (i >= K * N) return;
  int n = i / K, k = i - n * K;
  out[i] = f2bf(in[(size_t)k * N + n]);
}

// plain f32 -> bf16, 4 elems/thread, count divisible by 1024
__global__ __launch_bounds__(256) void k_conv_bf4(const float* __restrict__ in,
                                                  u16* __restrict__ out) {
  size_t e = ((size_t)blockIdx.x * 256 + threadIdx.x) * 4;
  float4 v = *(const float4*)(in + e);
  H4 o;
  o.u[0] = f2bf(v.x); o.u[1] = f2bf(v.y); o.u[2] = f2bf(v.z); o.u[3] = f2bf(v.w);
  *(uint2*)(out + e) = o.q;
}

// codebook squared norms from the bf16-rounded codebook (consistent with the GEMM)
__global__ __launch_bounds__(256) void k_cbnorm(const u16* __restrict__ cbbf,
                                                float* __restrict__ cbn) {
  int k = blockIdx.x * 256 + threadIdx.x;
  if (k >= KCB) return;
  float s = 0.f;
  #pragma unroll 8
  for (int c = 0; c < CH; ++c) { float v = bf2f(cbbf[k * CH + c]); s += v * v; }
  cbn[k] = s;
}

// tiled transpose per batch: in [B][R][Cc] -> out [B][Cc][R]
__global__ __launch_bounds__(256) void k_transpose(const float* __restrict__ in,
                                                   float* __restrict__ out, int R, int Cc) {
  __shared__ float tile[32][33];
  int b = blockIdx.z;
  int r0 = blockIdx.y * 32, c0 = blockIdx.x * 32;
  const float* ip = in  + (size_t)b * R * Cc;
  float*       op = out + (size_t)b * R * Cc;
  int tx = threadIdx.x, ty = threadIdx.y;
  for (int j = ty; j < 32; j += 8)
    tile[j][tx] = ip[(size_t)(r0 + j) * Cc + c0 + tx];
  __syncthreads();
  for (int j = ty; j < 32; j += 8)
    op[(size_t)(c0 + j) * R + r0 + tx] = tile[tx][j];
}

// ---------------- depthwise conv 7x7 + bias, NHWC in/out ----------------
__global__ __launch_bounds__(256) void k_dwconv(const float* __restrict__ X,
                                                const float* __restrict__ W,   // [C,49]
                                                const float* __restrict__ Bv,  // [C]
                                                float* __restrict__ Y) {
  const int n = blockIdx.x;            // b*4096 + s
  const int c = threadIdx.x;
  const int b = n >> 12, s = n & 4095;
  const int h = s >> 6, w = s & 63;
  float wr[49];
  #pragma unroll
  for (int t = 0; t < 49; ++t) wr[t] = W[c * 49 + t];
  float acc = Bv[c];
  #pragma unroll
  for (int kh = 0; kh < 7; ++kh) {
    int hh = h + kh - 3;
    if (hh < 0 || hh > 63) continue;
    #pragma unroll
    for (int kw = 0; kw < 7; ++kw) {
      int ww = w + kw - 3;
      if (ww < 0 || ww > 63) continue;
      acc += X[(size_t)(((b << 12) + (hh << 6) + ww)) * CH + c] * wr[kh * 7 + kw];
    }
  }
  Y[(size_t)n * CH + c] = acc;
}

// ---------------- LayerNorm over channels -> bf16 rows ----------------
__global__ __launch_bounds__(256) void k_layernorm(const float* __restrict__ X,
                                                   const float* __restrict__ G,
                                                   const float* __restrict__ Bt,
                                                   u16* __restrict__ O) {
  const int wave = threadIdx.x >> 5, lane = threadIdx.x & 31;
  const int row = blockIdx.x * 8 + wave;
  const float* rp = X + (size_t)row * CH + lane * 8;
  float x[8];
  float4 x0 = *(const float4*)rp, x1 = *(const float4*)(rp + 4);
  x[0]=x0.x; x[1]=x0.y; x[2]=x0.z; x[3]=x0.w;
  x[4]=x1.x; x[5]=x1.y; x[6]=x1.z; x[7]=x1.w;
  float s = 0.f, ss = 0.f;
  #pragma unroll
  for (int j = 0; j < 8; ++j) { s += x[j]; ss += x[j] * x[j]; }
  #pragma unroll
  for (int off = 1; off < 32; off <<= 1) {
    s  += __shfl_xor(s,  off, 32);
    ss += __shfl_xor(ss, off, 32);
  }
  float mu  = s * (1.0f / 256.0f);
  float var = fmaxf(ss * (1.0f / 256.0f) - mu * mu, 0.0f);
  float inv = rsqrtf(var + 1e-5f);
  H8 o;
  #pragma unroll
  for (int j = 0; j < 8; ++j) {
    int c = lane * 8 + j;
    o.u[j] = f2bf((x[j] - mu) * inv * G[c] + Bt[c]);
  }
  *(uint4*)(O + (size_t)row * CH + lane * 8) = o.q;
}

// ---------------- GEMM: C[M,N] = A[M,K](bf16) * BT[N,K](bf16)^T ----------------
// Ping-pong LDS double buffering; async VMEM->LDS staging; one barrier per k-step.
// EPI 1: +bias, exact GELU, store bf16, atomic sum-of-squares per (batch, col)
// EPI 2: +bias, +residual(in-place fp32 resIO)
template<int EPI>
__global__ __launch_bounds__(256)
void k_gemm(const u16* __restrict__ A, const u16* __restrict__ BT,
            const float* __restrict__ bias, u16* __restrict__ outBF,
            float* __restrict__ gxAcc, float* __restrict__ resIO,
            int M, int N, int K) {
  __shared__ u16 A_s[2][128 * PA];
  __shared__ u16 B_s[2][128 * PA];
  const int tid  = threadIdx.x;
  const int wave = tid >> 5, lane = tid & 31;
  const int m = lane & 15, half = lane >> 4;
  const int blockN = blockIdx.x * 128;
  const int blockM = blockIdx.y * 128;

  v8f acc[8];
  #pragma unroll
  for (int t = 0; t < 8; ++t) acc[t] = (v8f){0.f,0.f,0.f,0.f,0.f,0.f,0.f,0.f};

  const int lr = tid >> 1, lc = (tid & 1) * 16;
  const u16* gA = A  + (size_t)(blockM + lr) * K + lc;
  const u16* gB = BT + (size_t)(blockN + lr) * K + lc;
  const int KS = K >> 5;

  // prologue: stage k-step 0 into buffer 0
  stage32(gA, &A_s[0][lr * PA + lc]);
  stage32(gB, &B_s[0][lr * PA + lc]);

  for (int ks = 0; ks < KS; ++ks) {
    const int cur = ks & 1;
    stage_wait();
    __syncthreads();                      // tile ready; prev buffer's readers done
    if (ks + 1 < KS) {                    // prefetch next k-step into other buffer
      stage32(gA + ((ks + 1) << 5), &A_s[cur ^ 1][lr * PA + lc]);
      stage32(gB + ((ks + 1) << 5), &B_s[cur ^ 1][lr * PA + lc]);
    }
    Frag fa;
    const u16* rp = &A_s[cur][(wave * 16 + m) * PA];
    fa.q[0] = *(const uint4*)(rp + 8 * half);
    fa.q[1] = *(const uint4*)(rp + 8 * half + 16);
    #pragma unroll
    for (int nt = 0; nt < 8; ++nt) {
      Frag fb;
      const u16* rpb = &B_s[cur][(nt * 16 + m) * PA + 8 * half];
      fb.q[0] = *(const uint4*)(rpb);
      fb.q[1] = *(const uint4*)(rpb + 16);
      acc[nt] = __builtin_amdgcn_wmma_f32_16x16x32_bf16(false, fa.bf, false, fb.bf,
                                                        (short)0, acc[nt], false, false);
    }
  }

  #pragma unroll
  for (int nt = 0; nt < 8; ++nt) {
    const int colg = blockN + nt * 16 + m;
    const float bcol = bias[colg];
    if (EPI == 1) {
      float ssq = 0.f;
      #pragma unroll
      for (int v = 0; v < 8; ++v) {
        const int rowg = blockM + wave * 16 + 8 * half + v;
        float t = acc[nt][v] + bcol;
        float g = 0.5f * t * (1.0f + erff(t * 0.70710678118654752f));
        outBF[(size_t)rowg * N + colg] = f2bf(g);
        ssq += g * g;
      }
      ssq += __shfl_xor(ssq, 16, 32);
      if (half == 0) atomicAdd(&gxAcc[(blockM >> 12) * N + colg], ssq);
    } else {
      #pragma unroll
      for (int v = 0; v < 8; ++v) {
        const int rowg = blockM + wave * 16 + 8 * half + v;
        size_t off = (size_t)rowg * N + colg;
        resIO[off] = acc[nt][v] + bcol + resIO[off];
      }
    }
  }
}

// ---------------- GRN stage 2: gx = sqrt(ss); scale = 1 + gg * gx/(mean+eps) ----------------
__global__ __launch_bounds__(256) void k_grn2(const float* __restrict__ ssq,
                                              const float* __restrict__ gg,
                                              float* __restrict__ scale) {
  __shared__ float red[256];
  const int b = blockIdx.x, t = threadIdx.x;
  float g[4]; float s = 0.f;
  #pragma unroll
  for (int i = 0; i < 4; ++i) {
    g[i] = sqrtf(ssq[b * INTER + t + i * 256]);
    s += g[i];
  }
  red[t] = s; __syncthreads();
  for (int w = 128; w > 0; w >>= 1) { if (t < w) red[t] += red[t + w]; __syncthreads(); }
  float inv = 1.0f / (red[0] * (1.0f / (float)INTER) + 1e-6f);
  #pragma unroll
  for (int i = 0; i < 4; ++i)
    scale[b * INTER + t + i * 256] = 1.0f + gg[t + i * 256] * (g[i] * inv);
}

// ---------------- GRN apply, in place on bf16 activations ----------------
__global__ __launch_bounds__(256) void k_grn_apply(u16* __restrict__ h,
                                                   const float* __restrict__ scale,
                                                   const float* __restrict__ gb) {
  size_t e = ((size_t)blockIdx.x * 256 + threadIdx.x) * 4;
  int row = (int)(e >> 10), c = (int)(e & 1023), b = row >> 12;
  H4 io; io.q = *(uint2*)(h + e);
  #pragma unroll
  for (int j = 0; j < 4; ++j)
    io.u[j] = f2bf(bf2f(io.u[j]) * scale[b * INTER + c + j] + gb[c + j]);
  *(uint2*)(h + e) = io.q;
}

// ---------------- VQ: argmin_k (||cb_k||^2 - 2 f.cb_k) via WMMA ----------------
// 64 linear steps = 8 code-chunks x 8 k-steps, ping-pong staging across chunk
// boundaries, running first-min argmin in registers.
__global__ __launch_bounds__(256)
void k_vq(const u16* __restrict__ A, const u16* __restrict__ CB,
          const float* __restrict__ cbn, int* __restrict__ idxOut,
          float* __restrict__ idxOutF) {
  __shared__ u16 A_s[2][128 * PA];
  __shared__ u16 B_s[2][128 * PA];
  const int tid  = threadIdx.x;
  const int wave = tid >> 5, lane = tid & 31;
  const int m = lane & 15, half = lane >> 4;
  const int blockM = blockIdx.x * 128;
  const int lr = tid >> 1, lc = (tid & 1) * 16;

  float best[8]; int bidx[8];
  #pragma unroll
  for (int v = 0; v < 8; ++v) { best[v] = 3.4e38f; bidx[v] = 0; }

  const u16* gA = A + (size_t)(blockM + lr) * CH + lc;

  auto issue = [&](int s, int buf) {
    const int k0 = (s & 7) << 5;
    const int nc = (s >> 3) * 128;
    stage32(gA + k0, &A_s[buf][lr * PA + lc]);
    stage32(CB + (size_t)(nc + lr) * CH + k0 + lc, &B_s[buf][lr * PA + lc]);
  };

  v8f acc[8];
  #pragma unroll
  for (int t = 0; t < 8; ++t) acc[t] = (v8f){0.f,0.f,0.f,0.f,0.f,0.f,0.f,0.f};

  issue(0, 0);
  for (int s = 0; s < 64; ++s) {
    const int cur = s & 1;
    stage_wait();
    __syncthreads();
    if (s + 1 < 64) issue(s + 1, cur ^ 1);

    Frag fa;
    const u16* rp = &A_s[cur][(wave * 16 + m) * PA];
    fa.q[0] = *(const uint4*)(rp + 8 * half);
    fa.q[1] = *(const uint4*)(rp + 8 * half + 16);
    #pragma unroll
    for (int nt = 0; nt < 8; ++nt) {
      Frag fb;
      const u16* rpb = &B_s[cur][(nt * 16 + m) * PA + 8 * half];
      fb.q[0] = *(const uint4*)(rpb);
      fb.q[1] = *(const uint4*)(rpb + 16);
      acc[nt] = __builtin_amdgcn_wmma_f32_16x16x32_bf16(false, fa.bf, false, fb.bf,
                                                        (short)0, acc[nt], false, false);
    }

    if ((s & 7) == 7) {                    // end of a 128-code chunk
      const int nc = (s >> 3) * 128;
      #pragma unroll
      for (int nt = 0; nt < 8; ++nt) {
        const int colg = nc + nt * 16 + m;
        const float cn = cbn[colg];
        #pragma unroll
        for (int v = 0; v < 8; ++v) {
          float d = cn - 2.0f * acc[nt][v];
          if (d < best[v]) { best[v] = d; bidx[v] = colg; }  // cols increase -> first-min kept
        }
      }
      #pragma unroll
      for (int t = 0; t < 8; ++t) acc[t] = (v8f){0.f,0.f,0.f,0.f,0.f,0.f,0.f,0.f};
    }
  }

  // reduce across the 16 lanes of each half (same rows, different col sets)
  #pragma unroll
  for (int off = 8; off >= 1; off >>= 1) {
    #pragma unroll
    for (int v = 0; v < 8; ++v) {
      float ov = __shfl_xor(best[v], off, 32);
      int   oi = __shfl_xor(bidx[v], off, 32);
      if (ov < best[v] || (ov == best[v] && oi < bidx[v])) { best[v] = ov; bidx[v] = oi; }
    }
  }
  if (m == 0) {
    #pragma unroll
    for (int v = 0; v < 8; ++v) {
      int rowg = blockM + wave * 16 + 8 * half + v;
      idxOut[rowg]  = bidx[v];
      idxOutF[rowg] = (float)bidx[v];
    }
  }
}

// ---------------- gather codebook rows: NHWC decoder input + NCHW quantized out ----------------
__global__ __launch_bounds__(256) void k_gather(const int* __restrict__ idx,
                                                const float* __restrict__ cbF,
                                                float* __restrict__ Xdec,
                                                float* __restrict__ qNCHW) {
  const int n = blockIdx.x, c = threadIdx.x;
  const int k = idx[n];
  const int b = n >> 12, s = n & 4095;
  float q = cbF[(size_t)k * CH + c];
  Xdec[(size_t)n * CH + c] = q;
  qNCHW[((size_t)(b * CH + c) << 12) + s] = q;
}

// ---------------- host orchestration ----------------
extern "C" void kernel_launch(void* const* d_in, const int* in_sizes, int n_in,
                              void* d_out, int out_size, void* d_ws, size_t ws_size,
                              hipStream_t stream) {
  (void)in_sizes; (void)n_in; (void)out_size; (void)ws_size;
  const float* xin = (const float*)d_in[0];
  const float* cbF = (const float*)d_in[1];
  const float* P[20];
  for (int i = 0; i < 20; ++i) P[i] = (const float*)d_in[2 + i];
  // P layout per stack (enc=0..9, dec=10..19):
  // 0 conv_w 1 conv_b 2 ln_g 3 ln_b 4 fc1_w 5 fc1_b 6 grn_g 7 grn_b 8 fc2_w 9 fc2_b

  float* out_q   = (float*)d_out;                       // [8,256,64,64]
  float* out_dec = out_q + (size_t)NPOS * CH;           // [8,256,64,64]
  float* out_idx = out_dec + (size_t)NPOS * CH;         // [32768]

  // workspace carve-up
  float* X1    = (float*)d_ws;
  float* X2    = X1 + (size_t)NPOS * CH;
  float* T0    = X2 + (size_t)NPOS * CH;
  float* GX    = T0 + (size_t)NPOS * CH;
  float* SCL   = GX + BATCH * INTER;
  float* CBN   = SCL + BATCH * INTER;
  int*   IDX   = (int*)(CBN + KCB);
  u16*   ABF   = (u16*)(IDX + NPOS);
  u16*   HBF   = ABF + (size_t)NPOS * CH;
  u16*   W1T   = HBF + (size_t)NPOS * INTER;            // 4 * [1024,256]
  u16*   W2T   = W1T + 4 * (size_t)INTER * CH;          // 4 * [256,1024]
  u16*   CBBF  = W2T + 4 * (size_t)INTER * CH;          // [1024,256]

  // ---- weight prep (bf16, transposed to [N,K]) ----
  for (int blk = 0; blk < 4; ++blk) {
    int stack = blk >> 1, d = blk & 1;
    const float* w1 = P[stack * 10 + 4] + (size_t)d * CH * INTER;   // [256,1024]
    const float* w2 = P[stack * 10 + 8] + (size_t)d * INTER * CH;   // [1024,256]
    k_convT<<<(CH * INTER) / 256, 256, 0, stream>>>(w1, W1T + (size_t)blk * INTER * CH, CH, INTER);
    k_convT<<<(CH * INTER) / 256, 256, 0, stream>>>(w2, W2T + (size_t)blk * INTER * CH, INTER, CH);
  }
  k_conv_bf4<<<(KCB * CH) / 1024, 256, 0, stream>>>(cbF, CBBF);
  k_cbnorm<<<KCB / 256, 256, 0, stream>>>(CBBF, CBN);

  // ---- input NCHW -> NHWC ----
  k_transpose<<<dim3(HWS / 32, CH / 32, BATCH), dim3(32, 8), 0, stream>>>(xin, X1, CH, HWS);

  auto run_block = [&](float* Xio, int stack, int d) {
    int blk = stack * 2 + d;
    const float* cw  = P[stack * 10 + 0] + (size_t)d * CH * 49;
    const float* cbb = P[stack * 10 + 1] + (size_t)d * CH;
    const float* lng = P[stack * 10 + 2] + (size_t)d * CH;
    const float* lnb = P[stack * 10 + 3] + (size_t)d * CH;
    const float* b1  = P[stack * 10 + 5] + (size_t)d * INTER;
    const float* gg  = P[stack * 10 + 6] + (size_t)d * INTER;
    const float* gb  = P[stack * 10 + 7] + (size_t)d * INTER;
    const float* b2  = P[stack * 10 + 9] + (size_t)d * CH;
    const u16* w1t = W1T + (size_t)blk * INTER * CH;
    const u16* w2t = W2T + (size_t)blk * INTER * CH;

    k_dwconv<<<NPOS, 256, 0, stream>>>(Xio, cw, cbb, T0);
    k_layernorm<<<NPOS / 8, 256, 0, stream>>>(T0, lng, lnb, ABF);
    k_zero_f32<<<(BATCH * INTER) / 256, 256, 0, stream>>>(GX, BATCH * INTER);
    k_gemm<1><<<dim3(INTER / 128, NPOS / 128), 256, 0, stream>>>(
        ABF, w1t, b1, HBF, GX, nullptr, NPOS, INTER, CH);
    k_grn2<<<BATCH, 256, 0, stream>>>(GX, gg, SCL);
    k_grn_apply<<<((size_t)NPOS * INTER) / 1024, 256, 0, stream>>>(HBF, SCL, gb);
    k_gemm<2><<<dim3(CH / 128, NPOS / 128), 256, 0, stream>>>(
        HBF, w2t, b2, nullptr, nullptr, Xio, NPOS, CH, INTER);
  };

  // ---- encoder ----
  run_block(X1, 0, 0);
  run_block(X1, 0, 1);

  // ---- vector quantization ----
  k_conv_bf4<<<((size_t)NPOS * CH) / 1024, 256, 0, stream>>>(X1, ABF);
  k_vq<<<NPOS / 128, 256, 0, stream>>>(ABF, CBBF, CBN, IDX, out_idx);
  k_gather<<<NPOS, 256, 0, stream>>>(IDX, cbF, X2, out_q);

  // ---- decoder ----
  run_block(X2, 1, 0);
  run_block(X2, 1, 1);

  // ---- decoded NHWC -> NCHW ----
  k_transpose<<<dim3(CH / 32, HWS / 32, BATCH), dim3(32, 8), 0, stream>>>(X2, out_dec, HWS, CH);
}
